// SequenceBlock_31825707664111
// MI455X (gfx1250) — matrix operand order repaired and checked by reading, not compiled, not run on hardware
//
#include <hip/hip_runtime.h>
#include <math.h>

#define Bsz  4
#define Lseq 1024
#define Hd   512
#define Nst  64
#define NR   128      // real-augmented state dim (2*N)
#define TCH  128      // scan time-chunk
#define SROW 136      // padded per-column stride (ushorts) for state in LDS

typedef __attribute__((ext_vector_type(16))) __bf16        v16bf;
typedef __attribute__((ext_vector_type(8)))  float         v8f;
typedef __attribute__((ext_vector_type(4)))  unsigned int  u32x4;

union BfFrag { v16bf v; u32x4 q[2]; unsigned short s[16]; };

// Optional CDNA5 async global->LDS staging (ASYNCcnt path), guarded so the
// file still compiles if the toolchain lacks the builtins.
#if defined(__has_builtin)
# if __has_builtin(__builtin_amdgcn_global_load_async_to_lds_b128) && \
     __has_builtin(__builtin_amdgcn_s_wait_asynccnt)
#  define SCAN_ASYNC_LDS 1
# endif
#endif

#ifdef SCAN_ASYNC_LDS
// Builtin takes v4i (int __vector(4)) pointers per the clang prototype.
typedef __attribute__((__vector_size__(16))) int v4i;
typedef __attribute__((address_space(1))) v4i gv4i;   // global
typedef __attribute__((address_space(3))) v4i lv4i;   // LDS
#endif

__device__ __forceinline__ unsigned short f2bf(float f) {
    union { float f; unsigned u; } c; c.f = f;
    unsigned u = c.u;
    unsigned r = u + 0x7FFFu + ((u >> 16) & 1u);   // round-to-nearest-even
    return (unsigned short)(r >> 16);
}

// ---------------------------------------------------------------------------
// Kernel 1: DPLR -> discrete (Ab, Bb) via closed-form Woodbury expansion.
// Writes M_bf16[128][128] = [[Ar,-Ai],[Ai,Ar]], Bb_aug[128], c_aug[128].
// ---------------------------------------------------------------------------
__global__ void setup_dplr(const float* __restrict__ lre, const float* __restrict__ lim,
                           const float* __restrict__ pre, const float* __restrict__ pim,
                           const float* __restrict__ bre, const float* __restrict__ bim,
                           const float* __restrict__ cre, const float* __restrict__ cim,
                           const float* __restrict__ lstep,
                           unsigned short* __restrict__ Mb,
                           float* __restrict__ Bba, float* __restrict__ ca)
{
    __shared__ float sh[5];
    int tid = threadIdx.x;
    if (tid == 0) {
        float st = expf(lstep[0]);
        float g2 = 2.0f / st;
        float qr = 0.f, qi = 0.f, vbr = 0.f, vbi = 0.f;
        for (int k = 0; k < Nst; k++) {
            float dr = g2 - lre[k], di = -lim[k];
            float dn = dr*dr + di*di;
            float Dr = dr/dn, Di = -di/dn;                 // Dinv_k
            float vr = pre[k]*Dr + pim[k]*Di;              // v_k = conj(p_k)*Dinv_k
            float vi = pre[k]*Di - pim[k]*Dr;
            qr  += vr*pre[k] - vi*pim[k];                  // qcp = sum v_k p_k
            qi  += vr*pim[k] + vi*pre[k];
            vbr += vr*bre[k] - vi*bim[k];                  // vb = sum v_k b_k
            vbi += vr*bim[k] + vi*bre[k];
        }
        sh[0]=qr; sh[1]=qi; sh[2]=vbr; sh[3]=vbi; sh[4]=g2;
    }
    __syncthreads();
    float qr = sh[0], qi = sh[1], vbr = sh[2], vbi = sh[3], g2 = sh[4];
    float er = 1.f + qr, ei = qi;
    float ed = er*er + ei*ei;
    float ir = er/ed, ii = -ei/ed;                         // 1/(1+qcp)

    for (int idx = tid; idx < Nst*Nst; idx += blockDim.x) {
        int i = idx >> 6, j = idx & 63;
        float dr = g2 - lre[i], di = -lim[i];
        float dn = dr*dr + di*di;
        float Dr = dr/dn, Di = -di/dn;                     // Dinv_i
        float dpr = Dr*pre[i] - Di*pim[i];                 // Dinv_i * p_i
        float dpi = Dr*pim[i] + Di*pre[i];
        float wr = dpr*ir - dpi*ii;                        // w_i
        float wi = dpr*ii + dpi*ir;
        float qjr = pre[j], qji = -pim[j];                 // conj(p_j)
        float ejr = g2 - lre[j], eji = -lim[j];
        float ejn = ejr*ejr + eji*eji;
        float Djr = ejr/ejn, Dji = -eji/ejn;               // Dinv_j
        float vjr = qjr*Djr - qji*Dji;                     // v_j
        float vji = qjr*Dji + qji*Djr;
        float gjr = g2 + lre[j], gji = lim[j];             // g_j
        float abr, abi;
        abr = -(dpr*qjr - dpi*qji);                        // -Dinv_i p_i conj(p_j)
        abi = -(dpr*qji + dpi*qjr);
        float vgr = vjr*gjr - vji*gji;
        float vgi = vjr*gji + vji*gjr;
        abr -= wr*vgr - wi*vgi;                            // -w_i v_j g_j
        abi -= wr*vgi + wi*vgr;
        float wqr = wr*qr - wi*qi;
        float wqi = wr*qi + wi*qr;
        abr += wqr*qjr - wqi*qji;                          // +w_i qcp conj(p_j)
        abi += wqr*qji + wqi*qjr;
        if (i == j) {
            float gir = g2 + lre[i], gii = lim[i];
            abr += Dr*gir - Di*gii;
            abi += Dr*gii + Di*gir;
        }
        Mb[i*NR + j]              = f2bf(abr);
        Mb[i*NR + Nst + j]        = f2bf(-abi);
        Mb[(Nst+i)*NR + j]        = f2bf(abi);
        Mb[(Nst+i)*NR + Nst + j]  = f2bf(abr);
    }

    if (tid < Nst) {
        int i = tid;
        float dr = g2 - lre[i], di = -lim[i];
        float dn = dr*dr + di*di;
        float Dr = dr/dn, Di = -di/dn;
        float dpr = Dr*pre[i] - Di*pim[i];
        float dpi = Dr*pim[i] + Di*pre[i];
        float wr = dpr*ir - dpi*ii;
        float wi = dpr*ii + dpi*ir;
        float t1r = Dr*bre[i] - Di*bim[i];
        float t1i = Dr*bim[i] + Di*bre[i];
        float t2r = wr*vbr - wi*vbi;
        float t2i = wr*vbi + wi*vbr;
        Bba[i]       = 2.0f*(t1r - t2r);
        Bba[Nst + i] = 2.0f*(t1i - t2i);
        ca[i]        = cre[i];       // y = Re(c.s) = c_re.s_re - c_im.s_im
        ca[Nst + i]  = -cim[i];
    }
}

// ---------------------------------------------------------------------------
// Kernel 2: transpose+block-quantize out_w/out2_w into WMMA-B-fragment-linear
// bf16 layout: tile (nt=n/64, kt=k/32) stored contiguously, inner (n%64)*32+k%32.
// ---------------------------------------------------------------------------
__global__ void convert_weights(const float* __restrict__ w1, const float* __restrict__ w2,
                                unsigned short* __restrict__ wt1, unsigned short* __restrict__ wt2)
{
    int idx = blockIdx.x * blockDim.x + threadIdx.x;   // 2*512*512 threads
    int m   = idx >> 18;
    int rem = idx & 0x3FFFF;
    int n   = rem >> 9;
    int k   = rem & 511;
    const float* w = m ? w2 : w1;
    unsigned short* dst = m ? wt2 : wt1;
    int d = ((n >> 6) * 16 + (k >> 5)) * 2048 + (n & 63) * 32 + (k & 31);
    dst[d] = f2bf(w[n * 512 + k]);                     // B[k][n] = W[n][k]
}

// ---------------------------------------------------------------------------
// Kernel 3: LayerNorm over H (eps=1e-5). One block per (b,t) row.
// ---------------------------------------------------------------------------
__global__ void layernorm_k(const float* __restrict__ x, const float* __restrict__ w,
                            const float* __restrict__ b, float* __restrict__ xn)
{
    int row = blockIdx.x;
    size_t base = (size_t)row * Hd;
    int tid = threadIdx.x;
    float v0 = x[base + tid], v1 = x[base + tid + 256];
    float s = v0 + v1, ss = v0*v0 + v1*v1;
    #pragma unroll
    for (int o = 16; o > 0; o >>= 1) {
        s  += __shfl_xor(s,  o, 32);
        ss += __shfl_xor(ss, o, 32);
    }
    __shared__ float sred[8], ssred[8], stats[2];
    int wid = tid >> 5, lane = tid & 31;
    if (!lane) { sred[wid] = s; ssred[wid] = ss; }
    __syncthreads();
    if (!tid) {
        float S = 0.f, SS = 0.f;
        for (int i = 0; i < 8; i++) { S += sred[i]; SS += ssred[i]; }
        float mu  = S * (1.0f / Hd);
        float var = SS * (1.0f / Hd) - mu * mu;
        stats[0] = mu; stats[1] = rsqrtf(var + 1e-5f);
    }
    __syncthreads();
    float mu = stats[0], rinv = stats[1];
    xn[base + tid]       = (v0 - mu) * rinv * w[tid]       + b[tid];
    xn[base + tid + 256] = (v1 - mu) * rinv * w[tid + 256] + b[tid + 256];
}

// ---------------------------------------------------------------------------
// Kernel 4: the S4 recurrence. One block owns 16 (b,h) lanes; 8 waves each own
// a 16-row strip of the 128-dim augmented state. A-fragments of M live in
// registers for the whole sequence; state is double-buffered bf16 in LDS so
// B-fragments are single ds_load_b128's. Per timestep: two independent 2-deep
// WMMA chains (halved XDL critical path), state store issued before the
// y-readout so only the store gates the barrier, then one barrier.
// ---------------------------------------------------------------------------
__global__ void __launch_bounds__(256) scan_k(const float* __restrict__ xn,
                                              const float* __restrict__ dptr,
                                              const unsigned short* __restrict__ Mb,
                                              const float* __restrict__ Bba,
                                              const float* __restrict__ ca,
                                              unsigned short* __restrict__ gout)
{
    __shared__ __align__(16) unsigned short Ml[NR * NR];        // 32 KB constant M
    __shared__ __align__(16) unsigned short Sb[2 * 16 * SROW];  // dbl-buffered state
    __shared__ __align__(16) float ulds[TCH * 16];
    __shared__ float ylds[TCH * 16];
    __shared__ float Bbl[NR], cl[NR];

    int tid  = threadIdx.x;
    int w    = tid >> 5, lane = tid & 31;
    int col  = lane & 15, half = lane >> 4;

    int bh0  = blockIdx.x * 16;
    int bidx = bh0 / Hd, h0 = bh0 % Hd;
    const float* xp = xn + ((size_t)bidx * Lseq) * Hd + h0;
    unsigned short* gp = gout + ((size_t)bidx * Lseq) * Hd + h0;
    float dval = dptr[0];

    for (int i = tid; i < (NR * NR) / 8; i += 256)
        ((u32x4*)Ml)[i] = ((const u32x4*)Mb)[i];
    for (int i = tid; i < 2 * 16 * SROW; i += 256) Sb[i] = 0;
    for (int i = tid; i < NR; i += 256) { Bbl[i] = Bba[i]; cl[i] = ca[i]; }
    __syncthreads();

    // A fragments of M: rows [16w,16w+16), constant over time -> registers.
    BfFrag A[4];
    int arow = 16 * w + col;
    #pragma unroll
    for (int kt = 0; kt < 4; kt++) {
        A[kt].q[0] = *(const u32x4*)(&Ml[arow * NR + 32 * kt + 16 * half]);
        A[kt].q[1] = *(const u32x4*)(&Ml[arow * NR + 32 * kt + 16 * half + 8]);
    }
    float cw[8], bw[8];
    #pragma unroll
    for (int r = 0; r < 8; r++) {
        int rowi = 16 * w + 8 * half + r;
        cw[r] = cl[rowi]; bw[r] = Bbl[rowi];
    }
    v8f vzero;
    #pragma unroll
    for (int r = 0; r < 8; r++) vzero[r] = 0.f;

    int cur = 0;
    for (int t0 = 0; t0 < Lseq; t0 += TCH) {
#ifdef SCAN_ASYNC_LDS
        // stage u-chunk via async global->LDS (ASYNCcnt); zero y while in flight
        for (int i = tid; i < TCH * 4; i += 256) {          // 16B chunks
            int tt = i >> 2, part = i & 3;
            const float* src = &xp[(size_t)(t0 + tt) * Hd + part * 4];
            float* dst = &ulds[tt * 16 + part * 4];
            __builtin_amdgcn_global_load_async_to_lds_b128(
                (gv4i*)(unsigned long long)src,
                (lv4i*)(unsigned int)(unsigned long long)dst,
                0, 0);
        }
        for (int i = tid; i < TCH * 16; i += 256) ylds[i] = 0.0f;
        __builtin_amdgcn_s_wait_asynccnt(0);
#else
        for (int i = tid; i < TCH * 16; i += 256) {
            int tt = i >> 4, c2 = i & 15;
            ulds[i] = xp[(size_t)(t0 + tt) * Hd + c2];
            ylds[i] = 0.0f;
        }
#endif
        if (t0 + TCH < Lseq) __builtin_prefetch(&xp[(size_t)(t0 + TCH) * Hd], 0, 1);
        __syncthreads();

        for (int tt = 0; tt < TCH; tt++) {
            float u = ulds[tt * 16 + col];
            v8f c0;
            #pragma unroll
            for (int r = 0; r < 8; r++) c0[r] = u * bw[r];     // C = u * Bb
            const unsigned short* sc = &Sb[cur * (16 * SROW) + col * SROW];
            BfFrag B0, B1, B2, B3;
            B0.q[0] = *(const u32x4*)(&sc[16 * half]);
            B0.q[1] = *(const u32x4*)(&sc[16 * half + 8]);
            B1.q[0] = *(const u32x4*)(&sc[32 + 16 * half]);
            B1.q[1] = *(const u32x4*)(&sc[32 + 16 * half + 8]);
            B2.q[0] = *(const u32x4*)(&sc[64 + 16 * half]);
            B2.q[1] = *(const u32x4*)(&sc[64 + 16 * half + 8]);
            B3.q[0] = *(const u32x4*)(&sc[96 + 16 * half]);
            B3.q[1] = *(const u32x4*)(&sc[96 + 16 * half + 8]);

            // two independent 2-deep WMMA chains -> half the serial XDL depth
            v8f a0 = __builtin_amdgcn_wmma_f32_16x16x32_bf16(
                         false, A[0].v, false, B0.v, (short)0, c0,    false, false);
            v8f a1 = __builtin_amdgcn_wmma_f32_16x16x32_bf16(
                         false, A[2].v, false, B2.v, (short)0, vzero, false, false);
            a0 = __builtin_amdgcn_wmma_f32_16x16x32_bf16(
                         false, A[1].v, false, B1.v, (short)0, a0,    false, false);
            a1 = __builtin_amdgcn_wmma_f32_16x16x32_bf16(
                         false, A[3].v, false, B3.v, (short)0, a1,    false, false);

            float sn[8];
            #pragma unroll
            for (int r = 0; r < 8; r++) sn[r] = a0[r] + a1[r]; // s_new

            // store new state FIRST: only this store gates the next timestep
            union { unsigned short s[8]; u32x4 q; } P;
            #pragma unroll
            for (int r = 0; r < 8; r++) P.s[r] = f2bf(sn[r]);
            *(u32x4*)(&Sb[(cur ^ 1) * (16 * SROW) + col * SROW + 16 * w + 8 * half]) = P.q;

            // y-readout overlaps the store/barrier latency
            float yp = 0.f;
            #pragma unroll
            for (int r = 0; r < 8; r++) yp += cw[r] * sn[r];
            yp += __shfl_xor(yp, 16, 32);
            if (half == 0) atomicAdd(&ylds[tt * 16 + col], yp);

            __syncthreads();
            cur ^= 1;
        }

        // chunk epilogue: y = ys + d*u -> tanh-GELU -> bf16 g
        for (int i = tid; i < TCH * 16; i += 256) {
            int tt = i >> 4, c2 = i & 15;
            float y = ylds[i] + dval * ulds[i];
            float g = 0.5f * y * (1.0f + tanhf(0.7978845608f * (y + 0.044715f * y * y * y)));
            gp[(size_t)(t0 + tt) * Hd + c2] = f2bf(g);
        }
        __syncthreads();
    }
}

// ---------------------------------------------------------------------------
// Kernel 5: fused dual GEMM + sigmoid gate + residual:
//   out = skip + (g@W1^T + b1) * sigmoid(g@W2^T + b2)
// 64x64 tile per block, both products share the A tile; WMMA bf16.
// ---------------------------------------------------------------------------
__global__ void __launch_bounds__(256) out_gemm_k(
        const unsigned short* __restrict__ g,
        const unsigned short* __restrict__ wt1, const unsigned short* __restrict__ wt2,
        const float* __restrict__ b1, const float* __restrict__ b2,
        const float* __restrict__ skip, float* __restrict__ out)
{
    __shared__ __align__(16) unsigned short Al[64 * 48];
    __shared__ __align__(16) unsigned short B1l[64 * 48];
    __shared__ __align__(16) unsigned short B2l[64 * 48];

    int tid  = threadIdx.x;
    int w    = tid >> 5, lane = tid & 31;
    int half = lane >> 4, n16 = lane & 15;
    int wr   = w & 3, wc = w >> 2;
    int r0   = blockIdx.x * 64;
    int n0   = blockIdx.y * 64;

    v8f acc[2][2];
    #pragma unroll
    for (int ct = 0; ct < 2; ct++)
        #pragma unroll
        for (int m = 0; m < 2; m++)
            #pragma unroll
            for (int r = 0; r < 8; r++) acc[ct][m][r] = 0.f;

    int trow = tid >> 2, tchk = tid & 3;
    for (int k0 = 0; k0 < Hd; k0 += 32) {
        int kt = k0 >> 5;
        *(u32x4*)(&Al[trow * 48 + tchk * 8]) =
            *(const u32x4*)(&g[(size_t)(r0 + trow) * Hd + k0 + tchk * 8]);
        size_t bbase = (size_t)(blockIdx.y * 16 + kt) * 2048 + (size_t)tid * 8;
        *(u32x4*)(&B1l[trow * 48 + tchk * 8]) = *(const u32x4*)(&wt1[bbase]);
        *(u32x4*)(&B2l[trow * 48 + tchk * 8]) = *(const u32x4*)(&wt2[bbase]);
        __syncthreads();

        BfFrag Af;
        Af.q[0] = *(const u32x4*)(&Al[(16 * wr + n16) * 48 + 16 * half]);
        Af.q[1] = *(const u32x4*)(&Al[(16 * wr + n16) * 48 + 16 * half + 8]);
        #pragma unroll
        for (int ct = 0; ct < 2; ct++) {
            int cb = (32 * wc + 16 * ct + n16) * 48 + 16 * half;
            BfFrag Bf1, Bf2;
            Bf1.q[0] = *(const u32x4*)(&B1l[cb]);
            Bf1.q[1] = *(const u32x4*)(&B1l[cb + 8]);
            Bf2.q[0] = *(const u32x4*)(&B2l[cb]);
            Bf2.q[1] = *(const u32x4*)(&B2l[cb + 8]);
            acc[ct][0] = __builtin_amdgcn_wmma_f32_16x16x32_bf16(
                             false, Af.v, false, Bf1.v, (short)0, acc[ct][0], false, false);
            acc[ct][1] = __builtin_amdgcn_wmma_f32_16x16x32_bf16(
                             false, Af.v, false, Bf2.v, (short)0, acc[ct][1], false, false);
        }
        __syncthreads();
    }

    #pragma unroll
    for (int ct = 0; ct < 2; ct++) {
        int colg = n0 + 32 * wc + 16 * ct + n16;
        float bb1 = b1[colg], bb2 = b2[colg];
        #pragma unroll
        for (int r = 0; r < 8; r++) {
            int rowg = r0 + 16 * wr + 8 * half + r;
            size_t o = (size_t)rowg * Hd + colg;
            float z1 = acc[ct][0][r] + bb1;
            float z2 = acc[ct][1][r] + bb2;
            float gate = 1.0f / (1.0f + __expf(-z2));
            out[o] = skip[o] + z1 * gate;
        }
    }
}

// ---------------------------------------------------------------------------
extern "C" void kernel_launch(void* const* d_in, const int* in_sizes, int n_in,
                              void* d_out, int out_size, void* d_ws, size_t ws_size,
                              hipStream_t stream)
{
    const float* x       = (const float*)d_in[0];
    const float* nw      = (const float*)d_in[1];
    const float* nb      = (const float*)d_in[2];
    const float* lam_re  = (const float*)d_in[3];
    const float* lam_im  = (const float*)d_in[4];
    const float* p_re    = (const float*)d_in[5];
    const float* p_im    = (const float*)d_in[6];
    const float* b_re    = (const float*)d_in[7];
    const float* b_im    = (const float*)d_in[8];
    const float* c_re    = (const float*)d_in[9];
    const float* c_im    = (const float*)d_in[10];
    const float* dsc     = (const float*)d_in[11];
    const float* logstep = (const float*)d_in[12];
    const float* out_w   = (const float*)d_in[13];
    const float* out_b   = (const float*)d_in[14];
    const float* out2_w  = (const float*)d_in[15];
    const float* out2_b  = (const float*)d_in[16];

    // workspace layout (~13.7 MB total, 16B-aligned slices)
    char* ws = (char*)d_ws;
    unsigned short* Mb   = (unsigned short*)(ws + 0);                       // 32768 B
    float*          Bba  = (float*)(ws + 32768);                            // 512 B
    float*          caug = (float*)(ws + 33280);                            // 512 B
    float*          xn   = (float*)(ws + 65536);                            // 8 MB
    unsigned short* gbuf = (unsigned short*)(ws + 65536 + 8388608);         // 4 MB
    unsigned short* wt1  = (unsigned short*)(ws + 65536 + 8388608 + 4194304);
    unsigned short* wt2  = (unsigned short*)(ws + 65536 + 8388608 + 4194304 + 524288);

    setup_dplr<<<1, 256, 0, stream>>>(lam_re, lam_im, p_re, p_im, b_re, b_im,
                                      c_re, c_im, logstep, Mb, Bba, caug);
    convert_weights<<<2048, 256, 0, stream>>>(out_w, out2_w, wt1, wt2);
    layernorm_k<<<Bsz * Lseq, 256, 0, stream>>>(x, nw, nb, xn);
    scan_k<<<(Bsz * Hd) / 16, 256, 0, stream>>>(xn, dsc, Mb, Bba, caug, gbuf);
    out_gemm_k<<<dim3((Bsz * Lseq) / 64, Hd / 64), 256, 0, stream>>>(
        gbuf, wt1, wt2, out_b, out2_b, x, (float*)d_out);
}
